// BoundaryAwareMultiScaleFusion_73083163509095
// MI455X (gfx1250) — compile-verified, matrix operand order checked by reading, not run on hardware
//
// BoundaryAwareMultiScaleFusion for MI455X (gfx1250): wave32, bf16 WMMA, TDM staging.
#include <hip/hip_runtime.h>
#include <math.h>

#define BN 4
#define NPTS 4096
#define NPTOT (BN*NPTS)
#define RD 320
#define NCLS 17
#define KNN 12
#define NKEEP 13
#define TEMPR 0.75f

typedef __attribute__((ext_vector_type(16))) __bf16 v16bf;
typedef __attribute__((ext_vector_type(8)))  float  v8f;
typedef unsigned int u32x4 __attribute__((ext_vector_type(4)));
typedef int          i32x8 __attribute__((ext_vector_type(8)));
typedef int          i32x4 __attribute__((ext_vector_type(4)));

static __device__ __forceinline__ unsigned short f2bf(float x) {
  unsigned u = __float_as_uint(x);
  u += 0x7FFFu + ((u >> 16) & 1u);      // RNE
  return (unsigned short)(u >> 16);
}
static __device__ __forceinline__ unsigned pk2(float a, float b) {
  return (unsigned)f2bf(a) | ((unsigned)f2bf(b) << 16);
}

union Frag { uint4 u[2]; v16bf v; };
union Acc8 { v8f v; float f[8]; };

// --------------------------------------------------------------------------
// TDM: 2D tile load Global->LDS.  dims/strides in 8-byte units.
// LDS pad: +16B after every 128B stored  => LDS row pitch 144B (72 halves).
// --------------------------------------------------------------------------
static __device__ __forceinline__ void tdm_load2d(
    unsigned lds_off, const void* g, unsigned tensor_d0, unsigned tensor_d1,
    unsigned stride_d0, unsigned tile_d0, unsigned tile_d1)
{
  unsigned long long ga = (unsigned long long)g;
  u32x4 g0;
  g0[0] = 1u;                                          // count=1, user desc
  g0[1] = lds_off;                                     // lds_addr (bytes)
  g0[2] = (unsigned)ga;                                // global_addr[31:0]
  g0[3] = (unsigned)((ga >> 32) & 0x1FFFFFFull) | (2u << 30);  // addr[56:32] | type=2
  i32x8 g1;
  g1[0] = (int)((3u << 16)   // data_size = 8B
              | (1u << 20)   // pad_enable
              | (4u << 22)   // pad_interval: 32 dwords (128B)
              | (3u << 25)); // pad_amount: 4 dwords (16B)
  g1[1] = (int)(tensor_d0 << 16);
  g1[2] = (int)((tensor_d0 >> 16) | (tensor_d1 << 16));
  g1[3] = (int)((tensor_d1 >> 16) | (tile_d0 << 16));
  g1[4] = (int)(tile_d1 & 0xFFFFu);                    // tile_dim1 (tile_dim2=0)
  g1[5] = (int)stride_d0;                              // tensor_dim0_stride[31:0]
  g1[6] = 0;
  g1[7] = 0;
  i32x4 g2 = { 1, 1, 0, 0 };                           // tensor_dim2=1, tensor_dim3=1
  i32x4 g3 = { 0, (int)(1u << 16), 0, 0 };             // tensor_dim4=1
#if __clang_major__ >= 23
  i32x8 zz = { 0, 0, 0, 0, 0, 0, 0, 0 };
  __builtin_amdgcn_tensor_load_to_lds(g0, g1, g2, g3, zz, 0);
#else
  __builtin_amdgcn_tensor_load_to_lds(g0, g1, g2, g3, 0);
#endif
}

// ---------------- WMMA GEMM, bf16 in / fp32 acc, TDM-staged tiles ----------
// Block: 8 wave32s. Tile 256(M) x 64(N), K-chunk 64.
// Per wave: 32 rows x 64 cols = 2x4 accumulators; 8 WMMAs per 6 frag loads.
#define TM 256
#define TN 64
#define KC 64
#define LSTR2 72   // halves per LDS row (144B pitch)

__launch_bounds__(256)
__global__ void gemm_wmma(const unsigned short* __restrict__ A,   // [M,K] bf16 row-major
                          const unsigned short* __restrict__ Wt,  // [N,K] bf16 row-major (W^T)
                          const float* __restrict__ bias,
                          float* __restrict__ C, unsigned short* __restrict__ Cbf,
                          int M, int K, int N, int relu,
                          const float* __restrict__ res, int resStride)
{
  __shared__ unsigned short ldsA[2 * TM * LSTR2];   // 72 KB
  __shared__ unsigned short ldsB[2 * TN * LSTR2];   // 18 KB
  const int tid  = threadIdx.x;
  const int lane = tid & 31;
  const int wave = tid >> 5;
  const int m0 = blockIdx.x * TM;
  const int n0 = blockIdx.y * TN;
  const int r    = lane & 15;
  const int half = lane >> 4;

  Acc8 acc[2][4];
  #pragma unroll
  for (int mi = 0; mi < 2; ++mi)
    #pragma unroll
    for (int t = 0; t < 4; ++t)
      #pragma unroll
      for (int e = 0; e < 8; ++e) acc[mi][t].f[e] = 0.0f;

  const unsigned strideA = (unsigned)(K >> 2);   // 8B units per row
  const int nk = K >> 6;

  if (wave == 0) {
    tdm_load2d((unsigned)(size_t)(void*)&ldsA[0], &A[(size_t)m0 * K],
               strideA, (unsigned)M, strideA, 16u, (unsigned)TM);
    tdm_load2d((unsigned)(size_t)(void*)&ldsB[0], &Wt[(size_t)n0 * K],
               strideA, (unsigned)N, strideA, 16u, (unsigned)TN);
  }

  for (int t = 0; t < nk; ++t) {
    const int buf = t & 1;
    if (wave == 0) __builtin_amdgcn_s_wait_tensorcnt(0);
    __syncthreads();                       // tile t resident for all waves
    if (wave == 0 && (t + 1) < nk) {       // prefetch tile t+1 into other buffer
      const int kc = (t + 1) << 6;
      tdm_load2d((unsigned)(size_t)(void*)&ldsA[(buf ^ 1) * TM * LSTR2],
                 &A[(size_t)m0 * K + kc], strideA, (unsigned)M, strideA, 16u, (unsigned)TM);
      tdm_load2d((unsigned)(size_t)(void*)&ldsB[(buf ^ 1) * TN * LSTR2],
                 &Wt[(size_t)n0 * K + kc], strideA, (unsigned)N, strideA, 16u, (unsigned)TN);
    }
    const unsigned short* bA = &ldsA[buf * TM * LSTR2];
    const unsigned short* bB = &ldsB[buf * TN * LSTR2];
    #pragma unroll
    for (int ks = 0; ks < 2; ++ks) {
      // load ALL fragments for this k-subtile into distinct values first,
      // so the scheduler can overlap ds_loads with WMMA issue
      Frag fa[2], fb[4];
      #pragma unroll
      for (int mi = 0; mi < 2; ++mi) {
        const unsigned short* rA = &bA[(wave * 32 + mi * 16 + r) * LSTR2 + ks * 32 + half * 8];
        fa[mi].u[0] = *(const uint4*)rA;
        fa[mi].u[1] = *(const uint4*)(rA + 16);
      }
      #pragma unroll
      for (int nt = 0; nt < 4; ++nt) {
        const unsigned short* rB = &bB[(nt * 16 + r) * LSTR2 + ks * 32 + half * 8];
        fb[nt].u[0] = *(const uint4*)rB;
        fb[nt].u[1] = *(const uint4*)(rB + 16);
      }
      #pragma unroll
      for (int mi = 0; mi < 2; ++mi)
        #pragma unroll
        for (int nt = 0; nt < 4; ++nt)
          acc[mi][nt].v = __builtin_amdgcn_wmma_f32_16x16x32_bf16(
              false, fa[mi].v, false, fb[nt].v, (short)0, acc[mi][nt].v, false, false);
    }
  }

  // epilogue: bias (+relu) (+residual), fp32 and/or bf16 outputs
  #pragma unroll
  for (int mi = 0; mi < 2; ++mi) {
    #pragma unroll
    for (int nt = 0; nt < 4; ++nt) {
      int n = n0 + nt * 16 + r;
      float b = bias ? bias[n] : 0.0f;
      int mb = m0 + wave * 32 + mi * 16 + half * 8;
      #pragma unroll
      for (int v = 0; v < 8; ++v) {
        float val = acc[mi][nt].f[v] + b;
        if (relu) val = fmaxf(val, 0.0f);
        if (res)  val += res[(size_t)(mb + v) * resStride + n];
        if (C)    C[(size_t)(mb + v) * N + n] = val;
        if (Cbf)  Cbf[(size_t)(mb + v) * N + n] = f2bf(val);
      }
    }
  }
}

// ---------------- fp32 -> bf16 (8 elems/thread) ----------------------------
__global__ void tobf16_kernel(const float* __restrict__ src, unsigned short* __restrict__ dst,
                              int n8)
{
  int i = blockIdx.x * 256 + threadIdx.x;
  if (i >= n8) return;
  const float4* s = (const float4*)(src + (size_t)i * 8);
  float4 a = s[0], b = s[1];
  ((uint4*)dst)[i] = make_uint4(pk2(a.x, a.y), pk2(a.z, a.w), pk2(b.x, b.y), pk2(b.z, b.w));
}

// ---------------- W[K,N] fp32 -> W^T[N,Kpad] bf16 (zero-padded K) ----------
__global__ void wtrans_kernel(const float* __restrict__ W, unsigned short* __restrict__ Wt,
                              int K, int N, int Kpad)
{
  int idx = blockIdx.x * 256 + threadIdx.x;
  if (idx >= N * Kpad) return;
  int n = idx / Kpad, k = idx % Kpad;
  Wt[idx] = (k < K) ? f2bf(W[(size_t)k * N + n]) : (unsigned short)0;
}

// ---------------- kNN + boundary statistics --------------------------------
__global__ void boundary_kernel(const float* __restrict__ pos, const int* __restrict__ labels,
                                const float* __restrict__ logits, float* __restrict__ info)
{
  __shared__ float spx[256], spy[256], spz[256];
  const int tid = threadIdx.x;
  const int pid = blockIdx.x * 256 + tid;
  const int base = (pid >> 12) << 12;
  const float* myp = pos + (size_t)pid * 3;
  const float px = myp[0], py = myp[1], pz = myp[2];

  float d2v[NKEEP]; int idv[NKEEP];
  #pragma unroll
  for (int s = 0; s < NKEEP; ++s) { d2v[s] = 3.0e38f; idv[s] = 0; }

  for (int j0 = 0; j0 < NPTS; j0 += 256) {
    __syncthreads();
    const float* tp = pos + (size_t)(base + j0 + tid) * 3;
    spx[tid] = tp[0]; spy[tid] = tp[1]; spz[tid] = tp[2];
    __syncthreads();
    for (int jj = 0; jj < 256; ++jj) {
      float dx = spx[jj] - px, dy = spy[jj] - py, dz = spz[jj] - pz;
      float d2 = dx*dx + dy*dy + dz*dz;
      if (d2 < d2v[NKEEP-1]) {
        int j = j0 + jj;
        #pragma unroll
        for (int s = NKEEP-1; s >= 1; --s) {
          bool cp = d2 < d2v[s-1];
          bool cs = d2 < d2v[s];
          float nv = cp ? d2v[s-1] : (cs ? d2 : d2v[s]);
          int   ni = cp ? idv[s-1] : (cs ? j  : idv[s]);
          d2v[s] = nv; idv[s] = ni;
        }
        if (d2 < d2v[0]) { d2v[0] = d2; idv[0] = j; }
      }
    }
  }

  const int ol = labels[pid];
  float dist[KNN]; int df[KNN];
  #pragma unroll
  for (int k = 0; k < KNN; ++k) {
    dist[k] = sqrtf(d2v[k+1]);
    df[k] = (labels[base + idv[k+1]] != ol) ? 1 : 0;
  }
  float diffsum = 0.f, ssum = 0.f, sdist = 0.f, dsum = 0.f, bmin = 3.0e38f;
  #pragma unroll
  for (int k = 0; k < KNN; ++k) {
    diffsum += (float)df[k];
    if (df[k]) bmin = fminf(bmin, dist[k]);
    else { ssum += 1.f; sdist += dist[k]; }
    dsum += dist[k];
  }
  float dmean = dsum * (1.0f / KNN);
  float same_dist = sdist / (ssum + 1e-6f);
  float bdist = (diffsum > 0.f) ? bmin : same_dist;
  float var = 0.f;
  #pragma unroll
  for (int k = 0; k < KNN; ++k) { float d = dist[k] - dmean; var += d * d; }
  var *= (1.0f / (KNN - 1));
  float density = 1.0f / (dmean + 1e-6f);
  float curv = sqrtf(var) / (dmean + 1e-6f);

  const float* lg = logits + (size_t)pid * NCLS;
  float l[NCLS], mx = -3.0e38f;
  #pragma unroll
  for (int c = 0; c < NCLS; ++c) { l[c] = lg[c]; mx = fmaxf(mx, l[c]); }
  float S = 0.f;
  #pragma unroll
  for (int c = 0; c < NCLS; ++c) { l[c] = __expf((l[c] - mx) * (1.0f / TEMPR)); S += l[c]; }
  float inv = 1.0f / S, ent = 0.f;
  #pragma unroll
  for (int c = 0; c < NCLS; ++c) { float p = l[c] * inv; ent -= p * logf(p + 1e-8f); }
  ent *= (1.0f / logf((float)NCLS));

  float* o = info + (size_t)pid * 6;
  o[0] = diffsum * (1.0f / KNN); o[1] = inv; o[2] = ent;
  o[3] = density; o[4] = curv; o[5] = bdist;
}

// ---------------- boundary encoder 6->96->160 ------------------------------
__global__ void encoder_kernel(const float* __restrict__ info,
                               const float* __restrict__ W1, const float* __restrict__ b1,
                               const float* __restrict__ W2, const float* __restrict__ b2,
                               float* __restrict__ enc)
{
  __shared__ float sW1[6*96];
  __shared__ float sb1[96];
  __shared__ float sW2[96*160];
  __shared__ float sb2[160];
  const int tid = threadIdx.x;
  for (int i = tid; i < 6*96;   i += 256) sW1[i] = W1[i];
  for (int i = tid; i < 96;     i += 256) sb1[i] = b1[i];
  for (int i = tid; i < 96*160; i += 256) sW2[i] = W2[i];
  for (int i = tid; i < 160;    i += 256) sb2[i] = b2[i];
  __syncthreads();

  const int pid = blockIdx.x * 256 + tid;
  float x[6];
  #pragma unroll
  for (int c = 0; c < 6; ++c) x[c] = info[(size_t)pid * 6 + c];
  float h[96];
  for (int o = 0; o < 96; ++o) {
    float a = sb1[o];
    #pragma unroll
    for (int c = 0; c < 6; ++c) a += x[c] * sW1[c*96 + o];
    h[o] = fmaxf(a, 0.f);
  }
  float* ep = enc + (size_t)pid * 160;
  for (int o = 0; o < 160; ++o) {
    float a = sb2[o];
    for (int c = 0; c < 96; ++c) a += h[c] * sW2[c*160 + o];
    ep[o] = fmaxf(a, 0.f);
  }
}

// ---------------- attn_in(bf16,[N,512] zero-padded) + gf(fp32) -------------
__global__ void concat_kernel(const float* __restrict__ fp0, const float* __restrict__ fp1,
                              const float* __restrict__ fp2, const float* __restrict__ enc,
                              unsigned short* __restrict__ attn_b, float* __restrict__ gf)
{
  const int idx = blockIdx.x * 256 + threadIdx.x;   // NPTOT*512 threads
  const int p = idx >> 9, c = idx & 511;
  float v = 0.f;
  if (c < RD) {
    size_t o = (size_t)p * RD + c;
    v = (fp0[o] + fp1[o] + fp2[o]) * (1.0f / 3.0f);
    gf[o] = v;
  } else if (c < 480) {
    v = enc[(size_t)p * 160 + (c - RD)];
  }
  attn_b[idx] = f2bf(v);
}

// ---------------- attention head: ha[320] -> softmax over 3 scales ---------
__global__ void attn_kernel(const float* __restrict__ ha, const float* __restrict__ Wa2,
                            const float* __restrict__ ba2,
                            float* __restrict__ attn_ws, float* __restrict__ attn_out)
{
  __shared__ float sw[RD * 3];
  __shared__ float sb[3];
  const int tid = threadIdx.x;
  for (int i = tid; i < RD * 3; i += 256) sw[i] = Wa2[i];
  if (tid < 3) sb[tid] = ba2[tid];
  __syncthreads();

  const int pid = blockIdx.x * 256 + tid;
  const float* hp = ha + (size_t)pid * RD;
  float a0 = sb[0], a1 = sb[1], a2 = sb[2];
  for (int k = 0; k < RD; ++k) {
    float h = hp[k];
    a0 += h * sw[k*3 + 0];
    a1 += h * sw[k*3 + 1];
    a2 += h * sw[k*3 + 2];
  }
  float m = fmaxf(a0, fmaxf(a1, a2));
  float e0 = __expf(a0 - m), e1 = __expf(a1 - m), e2 = __expf(a2 - m);
  float inv = 1.0f / (e0 + e1 + e2);
  e0 *= inv; e1 *= inv; e2 *= inv;
  attn_ws[(size_t)pid*3 + 0] = e0; attn_ws[(size_t)pid*3 + 1] = e1; attn_ws[(size_t)pid*3 + 2] = e2;
  attn_out[(size_t)pid*3 + 0] = e0; attn_out[(size_t)pid*3 + 1] = e1; attn_out[(size_t)pid*3 + 2] = e2;
}

// ---------------- fused(bf16) = sum_s attn_s * fp_s ------------------------
__global__ void fuse_kernel(const float* __restrict__ fp0, const float* __restrict__ fp1,
                            const float* __restrict__ fp2, const float* __restrict__ attn,
                            unsigned short* __restrict__ fused_b)
{
  const int idx = blockIdx.x * 256 + threadIdx.x;   // NPTOT*RD threads
  const int p = idx / RD;
  float a0 = attn[(size_t)p*3 + 0], a1 = attn[(size_t)p*3 + 1], a2 = attn[(size_t)p*3 + 2];
  fused_b[idx] = f2bf(a0 * fp0[idx] + a1 * fp1[idx] + a2 * fp2[idx]);
}

// ---------------- driver ----------------------------------------------------
extern "C" void kernel_launch(void* const* d_in, const int* in_sizes, int n_in,
                              void* d_out, int out_size, void* d_ws, size_t ws_size,
                              hipStream_t stream)
{
  (void)in_sizes; (void)n_in; (void)out_size; (void)ws_size;
  const float* feat0 = (const float*)d_in[0];
  const float* feat1 = (const float*)d_in[1];
  const float* feat2 = (const float*)d_in[2];
  const float* logits= (const float*)d_in[3];
  const int*   labels= (const int*)  d_in[4];
  const float* pos   = (const float*)d_in[5];
  const float* Wp0 = (const float*)d_in[6];  const float* bp0 = (const float*)d_in[7];
  const float* Wp1 = (const float*)d_in[8];  const float* bp1 = (const float*)d_in[9];
  const float* Wp2 = (const float*)d_in[10]; const float* bp2 = (const float*)d_in[11];
  const float* Wbe1= (const float*)d_in[12]; const float* bbe1= (const float*)d_in[13];
  const float* Wbe2= (const float*)d_in[14]; const float* bbe2= (const float*)d_in[15];
  const float* Wa1 = (const float*)d_in[16]; const float* ba1 = (const float*)d_in[17];
  const float* Wa2 = (const float*)d_in[18]; const float* ba2 = (const float*)d_in[19];
  const float* Wo1 = (const float*)d_in[20]; const float* bo1 = (const float*)d_in[21];
  const float* Wo2 = (const float*)d_in[22]; const float* bo2 = (const float*)d_in[23];

  char* wsb = (char*)d_ws;
  size_t o = 0;
  auto alloc = [&](size_t bytes) { void* p = wsb + o; o += (bytes + 255) & ~(size_t)255; return p; };

  float* info   = (float*)alloc((size_t)NPTOT * 6   * 4);
  float* enc    = (float*)alloc((size_t)NPTOT * 160 * 4);
  float* fp0    = (float*)alloc((size_t)NPTOT * RD  * 4);
  float* fp1    = (float*)alloc((size_t)NPTOT * RD  * 4);
  float* fp2    = (float*)alloc((size_t)NPTOT * RD  * 4);
  float* gf     = (float*)alloc((size_t)NPTOT * RD  * 4);
  float* ha     = (float*)alloc((size_t)NPTOT * RD  * 4);
  float* attn   = (float*)alloc((size_t)NPTOT * 3   * 4);
  unsigned short* featb0 = (unsigned short*)alloc((size_t)NPTOT * 256 * 2);
  unsigned short* featb1 = (unsigned short*)alloc((size_t)NPTOT * 512 * 2);
  unsigned short* featb2 = (unsigned short*)alloc((size_t)NPTOT * 768 * 2);
  unsigned short* attnb  = (unsigned short*)alloc((size_t)NPTOT * 512 * 2);
  unsigned short* fusedb = (unsigned short*)alloc((size_t)NPTOT * RD  * 2);
  unsigned short* h1b    = (unsigned short*)alloc((size_t)NPTOT * RD  * 2);
  unsigned short* Wp0t = (unsigned short*)alloc((size_t)RD * 256 * 2);
  unsigned short* Wp1t = (unsigned short*)alloc((size_t)RD * 512 * 2);
  unsigned short* Wp2t = (unsigned short*)alloc((size_t)RD * 768 * 2);
  unsigned short* Wa1t = (unsigned short*)alloc((size_t)RD * 512 * 2);
  unsigned short* Wo1t = (unsigned short*)alloc((size_t)RD * RD  * 2);
  unsigned short* Wo2t = (unsigned short*)alloc((size_t)RD * RD  * 2);

  float* out_main = (float*)d_out;
  float* out_attn = out_main + (size_t)NPTOT * RD;

  // precision conversion / weight transposition (off the GEMM critical path)
  tobf16_kernel<<<(NPTOT*256/8+255)/256, 256, 0, stream>>>(feat0, featb0, NPTOT*256/8);
  tobf16_kernel<<<(NPTOT*512/8+255)/256, 256, 0, stream>>>(feat1, featb1, NPTOT*512/8);
  tobf16_kernel<<<(NPTOT*768/8+255)/256, 256, 0, stream>>>(feat2, featb2, NPTOT*768/8);
  wtrans_kernel<<<(RD*256+255)/256, 256, 0, stream>>>(Wp0, Wp0t, 256, RD, 256);
  wtrans_kernel<<<(RD*512+255)/256, 256, 0, stream>>>(Wp1, Wp1t, 512, RD, 512);
  wtrans_kernel<<<(RD*768+255)/256, 256, 0, stream>>>(Wp2, Wp2t, 768, RD, 768);
  wtrans_kernel<<<(RD*512+255)/256, 256, 0, stream>>>(Wa1, Wa1t, 480, RD, 512);
  wtrans_kernel<<<(RD*RD +255)/256, 256, 0, stream>>>(Wo1, Wo1t, RD, RD, RD);
  wtrans_kernel<<<(RD*RD +255)/256, 256, 0, stream>>>(Wo2, Wo2t, RD, RD, RD);

  boundary_kernel<<<NPTOT/256, 256, 0, stream>>>(pos, labels, logits, info);
  encoder_kernel<<<NPTOT/256, 256, 0, stream>>>(info, Wbe1, bbe1, Wbe2, bbe2, enc);

  dim3 gg(NPTOT / TM, RD / TN);   // 64 x 5
  gemm_wmma<<<gg, 256, 0, stream>>>(featb0, Wp0t, bp0, fp0, nullptr, NPTOT, 256, RD, 0, nullptr, 0);
  gemm_wmma<<<gg, 256, 0, stream>>>(featb1, Wp1t, bp1, fp1, nullptr, NPTOT, 512, RD, 0, nullptr, 0);
  gemm_wmma<<<gg, 256, 0, stream>>>(featb2, Wp2t, bp2, fp2, nullptr, NPTOT, 768, RD, 0, nullptr, 0);

  concat_kernel<<<(NPTOT*512)/256, 256, 0, stream>>>(fp0, fp1, fp2, enc, attnb, gf);
  gemm_wmma<<<gg, 256, 0, stream>>>(attnb, Wa1t, ba1, ha, nullptr, NPTOT, 512, RD, 1, nullptr, 0);
  attn_kernel<<<NPTOT/256, 256, 0, stream>>>(ha, Wa2, ba2, attn, out_attn);
  fuse_kernel<<<(NPTOT*RD)/256, 256, 0, stream>>>(fp0, fp1, fp2, attn, fusedb);

  gemm_wmma<<<gg, 256, 0, stream>>>(fusedb, Wo1t, bo1, nullptr, h1b, NPTOT, RD, RD, 1, nullptr, 0);
  gemm_wmma<<<gg, 256, 0, stream>>>(h1b, Wo2t, bo2, out_main, nullptr, NPTOT, RD, RD, 0, gf, RD);
}
